// HomologyGraphStats_89644557403168
// MI455X (gfx1250) — compile-verified, matrix operand order checked by reading, not run on hardware
//
#include <hip/hip_runtime.h>
#include <stdint.h>

#define NN 16384
#define DD 256
#define KK 15
#define NITER (NN / 64)

typedef __attribute__((ext_vector_type(16))) __bf16        v16bf;
typedef __attribute__((ext_vector_type(8)))  float         v8f;
typedef __attribute__((ext_vector_type(4)))  unsigned int  v4u;

union Frag { v4u u[2]; v16bf v; };

__device__ __forceinline__ unsigned short f2bf(float f) {
  unsigned int u = __float_as_uint(f);
  u += 0x7FFFu + ((u >> 16) & 1u);      // round-to-nearest-even
  return (unsigned short)(u >> 16);
}

// ---------------------------------------------------------------------------
// Kernel 1: L2-normalize rows (clamped at 1e-12 like F.normalize) -> bf16
// ---------------------------------------------------------------------------
__global__ void __launch_bounds__(256)
normalize_bf16(const float* __restrict__ x, unsigned short* __restrict__ xn) {
  __shared__ float red[256];
  const int row = blockIdx.x;
  const int t   = threadIdx.x;
  float v = x[(size_t)row * DD + t];
  red[t] = v * v;
  __syncthreads();
  #pragma unroll
  for (int s = 128; s > 0; s >>= 1) {
    if (t < s) red[t] += red[t + s];
    __syncthreads();
  }
  const float nrm   = sqrtf(red[0]);
  const float scale = 1.0f / fmaxf(nrm, 1e-12f);
  xn[(size_t)row * DD + t] = f2bf(v * scale);
}

// ---------------------------------------------------------------------------
// Kernel 2: fused  sim = Xn*Xn^T  (bf16 WMMA, fp32 acc)  +  streaming top-K
// Block: 128 threads (4 waves), owns 32 rows (two 16-row tiles). Each wave
// sweeps its 16-column lane of all 16384 columns. B fragments are double-
// buffered (ping-pong) and the first chunk of the next 64-column group is
// prefetched under the barriers / top-k phase.
// ---------------------------------------------------------------------------
__global__ void __launch_bounds__(128)
knn_topk(const unsigned short* __restrict__ xn, float* __restrict__ out) {
  __shared__ float simbuf[32 * 64];
  __shared__ float mv[32][4 * KK];
  __shared__ int   mi[32][4 * KK];

  const int t       = threadIdx.x;
  const int wave    = t >> 5;
  const int lane    = t & 31;
  const int g       = lane >> 4;     // K half-group (CDNA5 16-bit A/B layout)
  const int lrow    = lane & 15;     // matrix row (A) / col (B) within tile
  const int rowBase = blockIdx.x * 32;

  // ---- Preload A fragments for both 16-row tiles (VGPR-resident all sweep) --
  Frag a[2][8];
  #pragma unroll
  for (int h = 0; h < 2; ++h) {
    const unsigned short* abase = xn + (size_t)(rowBase + h * 16 + lrow) * DD;
    #pragma unroll
    for (int c = 0; c < 8; ++c) {
      a[h][c].u[0] = *reinterpret_cast<const v4u*>(abase + c * 32 + 8 * g);
      a[h][c].u[1] = *reinterpret_cast<const v4u*>(abase + c * 32 + 16 + 8 * g);
    }
  }

  // ---- Private sorted top-K (registers, fully unrolled accesses) ----
  float tv[KK]; int ti[KK];
  #pragma unroll
  for (int j = 0; j < KK; ++j) { tv[j] = -INFINITY; ti[j] = 0; }

  const int myrow = t >> 2;          // 0..31  (top-k phase mapping, 4 thr/row)
  const int sub   = t & 3;           // 0..3
  const int grow  = rowBase + myrow;
  const int rot   = (myrow * 4) & 63; // bank-conflict-free rotation

  // ---- B double buffer: prologue loads chunk 0 of iteration 0 ----
  const unsigned short* bbase = xn + (size_t)(wave * 16 + lrow) * DD;
  Frag b[2];
  b[0].u[0] = *reinterpret_cast<const v4u*>(bbase + 8 * g);
  b[0].u[1] = *reinterpret_cast<const v4u*>(bbase + 16 + 8 * g);

  for (int it = 0; it < NITER; ++it) {
    v8f acc0 = {0.f, 0.f, 0.f, 0.f, 0.f, 0.f, 0.f, 0.f};
    v8f acc1 = {0.f, 0.f, 0.f, 0.f, 0.f, 0.f, 0.f, 0.f};
    #pragma unroll
    for (int c = 0; c < 8; ++c) {
      // prefetch next chunk into the other buffer before consuming this one
      if (c < 7) {
        b[(c + 1) & 1].u[0] =
            *reinterpret_cast<const v4u*>(bbase + (c + 1) * 32 + 8 * g);
        b[(c + 1) & 1].u[1] =
            *reinterpret_cast<const v4u*>(bbase + (c + 1) * 32 + 16 + 8 * g);
      } else if (it + 1 < NITER) {   // chunk 0 of the next 64-column group
        b[0].u[0] = *reinterpret_cast<const v4u*>(bbase + 64 * DD + 8 * g);
        b[0].u[1] = *reinterpret_cast<const v4u*>(bbase + 64 * DD + 16 + 8 * g);
      }
      acc0 = __builtin_amdgcn_wmma_f32_16x16x32_bf16(
          false, a[0][c].v, false, b[c & 1].v, (short)0, acc0, false, false);
      acc1 = __builtin_amdgcn_wmma_f32_16x16x32_bf16(
          false, a[1][c].v, false, b[c & 1].v, (short)0, acc1, false, false);
    }
    bbase += 64 * DD;

    __syncthreads();   // previous chunk fully consumed
    #pragma unroll
    for (int r = 0; r < 8; ++r) {
      // C layout: VGPR r -> row r + 8*g, column lane&15
      simbuf[(r + 8 * g) * 64 + wave * 16 + lrow]      = acc0[r];
      simbuf[(16 + r + 8 * g) * 64 + wave * 16 + lrow] = acc1[r];
    }
    __syncthreads();   // tile visible to all threads

    #pragma unroll
    for (int q = 0; q < 16; ++q) {
      const int lcol = (sub + 4 * q + rot) & 63;   // rotated: 32 distinct banks
      const int gcol = it * 64 + lcol;
      float cand = simbuf[myrow * 64 + lcol];
      if (gcol == grow) cand = 0.0f;               // fill_diagonal_(0.0)
      if (cand > tv[KK - 1]) {                     // strict > : lower idx wins
        tv[KK - 1] = cand; ti[KK - 1] = gcol;
        #pragma unroll
        for (int j = KK - 1; j > 0; --j) {
          if (tv[j] > tv[j - 1]) {
            float tf = tv[j]; tv[j] = tv[j - 1]; tv[j - 1] = tf;
            int   tn = ti[j]; ti[j] = ti[j - 1]; ti[j - 1] = tn;
          }
        }
      }
    }
  }

  // ---- Merge the 4 partial lists per row, emit edges ----
  #pragma unroll
  for (int j = 0; j < KK; ++j) {
    mv[myrow][sub * KK + j] = tv[j];
    mi[myrow][sub * KK + j] = ti[j];
  }
  __syncthreads();

  const size_t NKt = (size_t)NN * KK;
  const size_t NEt = (size_t)NN * (KK + 1);
  if (t < 32) {
    const int r     = t;
    const int growr = rowBase + r;
    for (int k = 0; k < KK; ++k) {
      float best = -INFINITY; int bi = 0x7fffffff; int bp = 0;
      for (int p = 0; p < 4 * KK; ++p) {
        const float v  = mv[r][p];
        const int   id = mi[r][p];
        if (v > best || (v == best && id < bi)) { best = v; bi = id; bp = p; }
      }
      mv[r][bp] = -INFINITY;
      const size_t e = (size_t)growr * KK + k;
      out[e]           = (float)bi;      // edge_index[0] : neighbor (source)
      out[NEt + e]     = (float)growr;   // edge_index[1] : target row
      out[2 * NEt + e] = best;           // edge_attr     : similarity
    }
    // self loops appended after the N*K knn edges
    out[NKt + growr]           = (float)growr;
    out[NEt + NKt + growr]     = (float)growr;
    out[2 * NEt + NKt + growr] = 1.0f;
  }
}

// ---------------------------------------------------------------------------
extern "C" void kernel_launch(void* const* d_in, const int* in_sizes, int n_in,
                              void* d_out, int out_size, void* d_ws, size_t ws_size,
                              hipStream_t stream) {
  (void)in_sizes; (void)n_in; (void)out_size; (void)ws_size;
  const float* x  = (const float*)d_in[0];
  float* out      = (float*)d_out;
  unsigned short* xn = (unsigned short*)d_ws;   // 16384*256*2 = 8 MB scratch

  normalize_bf16<<<NN, 256, 0, stream>>>(x, xn);
  knn_topk<<<NN / 32, 128, 0, stream>>>(xn, out);
}
